// Model_31026843746806
// MI455X (gfx1250) — matrix operand order repaired and checked by reading, not compiled, not run on hardware
//
#include <hip/hip_runtime.h>
#include <hip/hip_bf16.h>
#include <cstdint>

typedef _Float16 v16h __attribute__((ext_vector_type(16)));
typedef float    v8f  __attribute__((ext_vector_type(8)));
typedef _Float16 h8   __attribute__((ext_vector_type(8)));

#define ENC 256
#define DI 512
#define NST 16
#define RLOW 16
#define KC 4
#define TT 1024
#define BB 4
#define ROWS (BB*TT)   // 4096

// ---------------- generic helpers ----------------

__global__ void cvt_f32_to_f16(const float* __restrict__ s, _Float16* __restrict__ d, int n) {
    int i = blockIdx.x * blockDim.x + threadIdx.x;
    if (i < n) d[i] = (_Float16)s[i];
}

__device__ __forceinline__ float blockSum256(float v, float* red) {
    int tid = threadIdx.x;
    red[tid] = v; __syncthreads();
    for (int off = 128; off; off >>= 1) {
        if (tid < off) red[tid] += red[tid + off];
        __syncthreads();
    }
    float r = red[0]; __syncthreads();
    return r;
}

// LayerNorm over last dim, f32 in -> f16 out.  blockDim must be power of 2 <= 256.
__global__ void ln_to_f16(const float* __restrict__ x, const float* __restrict__ g,
                          const float* __restrict__ b, _Float16* __restrict__ out, int width) {
    int row = blockIdx.x;
    const float* xr = x + (size_t)row * width;
    __shared__ float red[256];
    int tid = threadIdx.x, nt = blockDim.x;
    float s = 0.f;
    for (int i = tid; i < width; i += nt) s += xr[i];
    red[tid] = s; __syncthreads();
    for (int off = nt >> 1; off; off >>= 1) { if (tid < off) red[tid] += red[tid+off]; __syncthreads(); }
    float mean = red[0] / (float)width; __syncthreads();
    float v = 0.f;
    for (int i = tid; i < width; i += nt) { float d = xr[i] - mean; v += d * d; }
    red[tid] = v; __syncthreads();
    for (int off = nt >> 1; off; off >>= 1) { if (tid < off) red[tid] += red[tid+off]; __syncthreads(); }
    float rstd = rsqrtf(red[0] / (float)width + 1e-5f);
    for (int i = tid; i < width; i += nt)
        out[(size_t)row * width + i] = (_Float16)((xr[i] - mean) * rstd * g[i] + b[i]);
}

// ---------------- WMMA GEMM with async-to-LDS staged B ----------------
//  out[M x N] = A(M x K f16) @ W^T   (W is N x K f16 row-major)
//  Block = 8 waves = 128(M) x 16(N) strip; the 16xK weight panel is streamed
//  through a double-buffered LDS tile via GLOBAL_LOAD_ASYNC_TO_LDS_B128
//  (ASYNCcnt), shared by all 8 waves.  A fragments are register-pipelined
//  one K-step ahead so global latency overlaps the WMMA + barrier.
// modes
#define GM_F32       0   // out32 = c + bias
#define GM_F32_ACC   1   // out32 += c
#define GM_F16_GELU  2   // out16 = gelu(c + bias)
#define GM_F32_ADD   3   // out32 = c + bias + addend

__global__ __launch_bounds__(256) void wmma_gemm16(
    const _Float16* __restrict__ A, const _Float16* __restrict__ W,
    int M, int Nn, int K,
    float* __restrict__ out32, _Float16* __restrict__ out16, int ldc,
    const float* __restrict__ bias, const float* __restrict__ addend, int mode)
{
    __shared__ _Float16 bs[2][16 * 32];   // double-buffered 16(N) x 32(K) weight chunk

    int nb_n = Nn >> 4;
    int bm = blockIdx.x / nb_n;           // 128-row strip index
    int ntile = blockIdx.x - bm * nb_n;
    int wv = threadIdx.x >> 5;
    int mtile = bm * 8 + wv;
    int lane = threadIdx.x & 31;
    int l16 = lane & 15;
    int hi = lane >> 4;                   // 0: lanes 0-15, 1: lanes 16-31

    const _Float16* Arow = A + (size_t)(mtile * 16 + l16) * K;

    // async loader: waves 0-1 (threads 0-63), each lane moves 16B per chunk.
    // readfirstlane makes the guard an SGPR condition (no exec masking).
    bool loaderWave = __builtin_amdgcn_readfirstlane((int)threadIdx.x) < 64;
    int tid = threadIdx.x;
    int lrow = tid >> 2;                  // 0..15  (N row within tile)
    int lq = tid & 3;                     // 0..3   (8-half quarter of the 32-K row)
    const _Float16* Wbase = W + (size_t)(ntile * 16 + lrow) * K + lq * 8;
    unsigned ldsoff0 = (unsigned)(uintptr_t)&bs[0][lrow * 32 + lq * 8];
    unsigned ldsoff1 = (unsigned)(uintptr_t)&bs[1][lrow * 32 + lq * 8];

    if (loaderWave) {
        asm volatile("global_load_async_to_lds_b128 %0, %1, off"
                     :: "v"(ldsoff0), "v"(Wbase) : "memory");
    }

    const _Float16* bfrag = &bs[0][l16 * 32 + hi * 8];
    // preload A fragment for step 0
    h8 a0 = *(const h8*)(Arow + (hi ? 8 : 0));
    h8 a1 = *(const h8*)(Arow + 16 + (hi ? 8 : 0));

    v8f c = {};
    int nk = K >> 5;
    for (int step = 0; step < nk; ++step) {
        int cur = step & 1;
        // scalar wait: loader waves wait their async chunk; others have ASYNCcnt==0
        asm volatile("s_wait_asynccnt 0" ::: "memory");
        __syncthreads();   // publish chunk `cur`; prior reads of other buffer drained
        if (loaderWave) {
            if (step + 1 < nk) {
                const _Float16* gp = Wbase + ((size_t)(step + 1) << 5);
                unsigned dst = cur ? ldsoff0 : ldsoff1;
                asm volatile("global_load_async_to_lds_b128 %0, %1, off"
                             :: "v"(dst), "v"(gp) : "memory");
            }
        }
        // B fragment from LDS: N = l16 ; lanes 0-15: K 0..7,16..23 ; hi lanes: K 8..15,24..31
        const _Float16* bp = bfrag + cur * (16 * 32);
        h8 b0 = *(const h8*)(bp);
        h8 b1 = *(const h8*)(bp + 16);
        // prefetch next step's A fragment (clamped on the last iteration)
        int kn = ((step + 1 < nk) ? (step + 1) : step) << 5;
        h8 n0 = *(const h8*)(Arow + kn + (hi ? 8 : 0));
        h8 n1 = *(const h8*)(Arow + kn + 16 + (hi ? 8 : 0));
        v16h a, bf;
#pragma unroll
        for (int j = 0; j < 8; ++j) { a[j] = a0[j]; a[8+j] = a1[j]; bf[j] = b0[j]; bf[8+j] = b1[j]; }
        c = __builtin_amdgcn_wmma_f32_16x16x32_f16(false, a, false, bf, (short)0, c, false, false);
        a0 = n0; a1 = n1;
    }

    int n = ntile * 16 + l16;
    float bval = bias ? bias[n] : 0.0f;
#pragma unroll
    for (int j = 0; j < 8; ++j) {
        int m = mtile * 16 + j + (hi ? 8 : 0);
        size_t idx = (size_t)m * ldc + n;
        float v = c[j] + bval;
        if (mode == GM_F16_GELU) {
            out16[idx] = (_Float16)(0.5f * v * (1.0f + erff(v * 0.70710678118f)));
        } else if (mode == GM_F32_ACC) {
            out32[idx] += v;
        } else if (mode == GM_F32_ADD) {
            out32[idx] = v + addend[idx];
        } else {
            out32[idx] = v;
        }
    }
}

// ---------------- fusion front-end ----------------

// hcat[r][0:256]=f16(fw) ; hcat[r][256:512]=f16(resize(fs_small)) (half-pixel linear 512->1024)
__global__ void hcat_kernel(const float* __restrict__ fw, const float* __restrict__ fs,
                            _Float16* __restrict__ hcat) {
    int r = blockIdx.x; int e = threadIdx.x;
    int b = r >> 10; int t = r & 1023;
    hcat[(size_t)r * 512 + e] = (_Float16)fw[(size_t)r * 256 + e];
    float src = (t + 0.5f) * 0.5f - 0.5f;
    float fl = floorf(src);
    int i0 = (int)fl;
    float fr = src - fl;
    int ia = i0 < 0 ? 0 : (i0 > 511 ? 511 : i0);
    int i1 = i0 + 1;
    int ib = i1 < 0 ? 0 : (i1 > 511 ? 511 : i1);
    float v = (1.0f - fr) * fs[((size_t)(b * 512 + ia)) * 256 + e]
            +         fr  * fs[((size_t)(b * 512 + ib)) * 256 + e];
    hcat[(size_t)r * 512 + 256 + e] = (_Float16)v;
}

// SE gate: se[b][e] = sigmoid( relu(mean_t(fc) @ W1^T) @ W2^T )
__global__ void se_kernel(const float* __restrict__ fc, const float* __restrict__ W1,
                          const float* __restrict__ W2, float* __restrict__ se) {
    int b = blockIdx.x, e = threadIdx.x;
    const float* base = fc + (size_t)b * TT * 256;
    float s = 0.f;
    for (int t = 0; t < TT; ++t) s += base[(size_t)t * 256 + e];
    __shared__ float mean[256];
    __shared__ float hid[16];
    mean[e] = s * (1.0f / (float)TT);
    __syncthreads();
    if (e < 16) {
        float h = 0.f;
        for (int j = 0; j < 256; ++j) h += mean[j] * W1[e * 256 + j];
        hid[e] = h > 0.f ? h : 0.f;
    }
    __syncthreads();
    float a = 0.f;
#pragma unroll
    for (int j = 0; j < 16; ++j) a += hid[j] * W2[e * 16 + j];
    se[b * 256 + e] = 1.0f / (1.0f + expf(-a));
}

// fc*se -> LN(ln_f) -> res (f32) -> LN(n1) -> xn (f16)
__global__ void fuse_se_ln2_kernel(const float* __restrict__ fc, const float* __restrict__ se,
    const float* __restrict__ lfg, const float* __restrict__ lfb,
    const float* __restrict__ n1g, const float* __restrict__ n1b,
    float* __restrict__ res, _Float16* __restrict__ xn) {
    int r = blockIdx.x; int e = threadIdx.x; int b = r >> 10;
    __shared__ float red[256];
    float v = fc[(size_t)r * 256 + e] * se[b * 256 + e];
    float mean = blockSum256(v, red) * (1.0f / 256.0f);
    float dd = v - mean;
    float var = blockSum256(dd * dd, red) * (1.0f / 256.0f);
    float x = dd * rsqrtf(var + 1e-5f) * lfg[e] + lfb[e];
    res[(size_t)r * 256 + e] = x;
    float m2 = blockSum256(x, red) * (1.0f / 256.0f);
    float d2 = x - m2;
    float v2 = blockSum256(d2 * d2, red) * (1.0f / 256.0f);
    xn[(size_t)r * 256 + e] = (_Float16)(d2 * rsqrtf(v2 + 1e-5f) * n1g[e] + n1b[e]);
}

// ---------------- Mamba pieces ----------------

// Causal depthwise conv (KC=4) + SiLU over xm = xz[:, :512]; dir=blockIdx.z flips time.
__global__ void conv_silu_kernel(const float* __restrict__ xz, const float* __restrict__ cw,
                                 const float* __restrict__ cb, _Float16* __restrict__ xc) {
    int r = blockIdx.x; int b = r >> 10; int tl = r & 1023;
    int d = blockIdx.y * 256 + threadIdx.x;
    int dir = blockIdx.z;
    float acc = cb[d];
#pragma unroll
    for (int k = 0; k < KC; ++k) {
        int ts = tl + k - (KC - 1);
        if (ts >= 0) {
            int tp = dir ? (TT - 1 - ts) : ts;
            acc += xz[((size_t)(b * TT + tp)) * 1024 + d] * cw[d * KC + k];
        }
    }
    float y = acc / (1.0f + expf(-acc));
    xc[(size_t)dir * ROWS * DI + (size_t)r * DI + d] = (_Float16)y;
}

// Fused dt-projection + softplus + selective scan + u*D + silu(z) gate.
// grid (DI/128, B, 2 dirs), block 128: each thread owns channel d with 16 states.
__global__ __launch_bounds__(128) void scan_kernel(
    const float* __restrict__ xdbl, const _Float16* __restrict__ xc,
    const float* __restrict__ xz, const float* __restrict__ Wdt_g,
    const float* __restrict__ bdt_g, const float* __restrict__ Alog,
    const float* __restrict__ Dp, _Float16* __restrict__ g) {
    int tid = threadIdx.x;
    int d = blockIdx.x * 128 + tid;
    int b = blockIdx.y;
    int dir = blockIdx.z;
    const float*    xdb = xdbl + (size_t)dir * ROWS * 48;
    const _Float16* xcd = xc   + (size_t)dir * ROWS * DI;
    _Float16*       gd  = g    + (size_t)dir * ROWS * DI;
    float A_[NST], Wdt[NST], h[NST];
#pragma unroll
    for (int n = 0; n < NST; ++n) {
        A_[n] = -expf(Alog[d * NST + n]);
        Wdt[n] = Wdt_g[d * NST + n];
        h[n] = 0.0f;
    }
    float bdt = bdt_g[d], Dd = Dp[d];
    __shared__ float srow[48];
    for (int tl = 0; tl < TT; ++tl) {
        size_t r = (size_t)b * TT + tl;
        __syncthreads();
        if (tid < 48) srow[tid] = xdb[r * 48 + tid];
        __syncthreads();
        float u = (float)xcd[r * DI + d];
        float dtr = bdt;
#pragma unroll
        for (int j = 0; j < RLOW; ++j) dtr += srow[j] * Wdt[j];
        float dtv = dtr > 20.0f ? dtr : log1pf(expf(dtr));
        float y = 0.0f;
#pragma unroll
        for (int n = 0; n < NST; ++n) {
            float dA = expf(dtv * A_[n]);
            h[n] = dA * h[n] + (dtv * srow[16 + n]) * u;
            y += h[n] * srow[32 + n];
        }
        float yo = y + u * Dd;
        int tp = dir ? (TT - 1 - tl) : tl;
        size_t rp = (size_t)b * TT + tp;
        float zv = xz[rp * 1024 + DI + d];
        gd[rp * DI + d] = (_Float16)(yo * (zv / (1.0f + expf(-zv))));
    }
}

// ---------------- host ----------------

extern "C" void kernel_launch(void* const* d_in, const int* in_sizes, int n_in,
                              void* d_out, int out_size, void* d_ws, size_t ws_size,
                              hipStream_t stream) {
    const float* f_wavlm = (const float*)d_in[0];
    const float* f_sinc  = (const float*)d_in[1];
    const float* ln_w_g  = (const float*)d_in[2];
    const float* ln_w_b  = (const float*)d_in[3];
    const float* ln_s_g  = (const float*)d_in[4];
    const float* ln_s_b  = (const float*)d_in[5];
    const float* W_wp    = (const float*)d_in[6];
    const float* b_wp    = (const float*)d_in[7];
    const float* W_sp    = (const float*)d_in[8];
    const float* b_sp    = (const float*)d_in[9];
    const float* W_fuse  = (const float*)d_in[10];
    const float* b_fuse  = (const float*)d_in[11];
    const float* W_se1   = (const float*)d_in[12];
    const float* W_se2   = (const float*)d_in[13];
    const float* ln_f_g  = (const float*)d_in[14];
    const float* ln_f_b  = (const float*)d_in[15];
    const float* n1_g    = (const float*)d_in[16];
    const float* n1_b    = (const float*)d_in[17];
    const float* n2_g    = (const float*)d_in[18];
    const float* n2_b    = (const float*)d_in[19];
    const float* W_in    = (const float*)d_in[20];
    const float* conv_w  = (const float*)d_in[21];
    const float* conv_b  = (const float*)d_in[22];
    const float* W_xproj = (const float*)d_in[23];
    const float* W_dt    = (const float*)d_in[24];
    const float* b_dt    = (const float*)d_in[25];
    const float* A_log   = (const float*)d_in[26];
    const float* Dp      = (const float*)d_in[27];
    const float* W_out   = (const float*)d_in[28];
    const float* W_ff1   = (const float*)d_in[29];
    const float* b_ff1   = (const float*)d_in[30];
    const float* W_ff2   = (const float*)d_in[31];
    const float* b_ff2   = (const float*)d_in[32];
    (void)in_sizes; (void)n_in; (void)out_size; (void)ws_size;

    char* ws = (char*)d_ws;
    size_t off = 0;
    auto alloc = [&](size_t elems, size_t esz) -> void* {
        off = (off + 255) & ~(size_t)255;
        void* p = ws + off;
        off += elems * esz;
        return p;
    };

    _Float16* WHwp    = (_Float16*)alloc(256 * 1024, 2);
    _Float16* WHsp    = (_Float16*)alloc(256 * 64, 2);
    _Float16* WHfuse  = (_Float16*)alloc(256 * 512, 2);
    _Float16* WHin    = (_Float16*)alloc(1024 * 256, 2);
    _Float16* WHxproj = (_Float16*)alloc(48 * 512, 2);
    _Float16* WHout   = (_Float16*)alloc(256 * 512, 2);
    _Float16* WHff1   = (_Float16*)alloc(1024 * 256, 2);
    _Float16* WHff2   = (_Float16*)alloc(256 * 1024, 2);

    _Float16* HW   = (_Float16*)alloc((size_t)ROWS * 1024, 2);  // LN(wavlm); reused as FFN hidden
    _Float16* HS   = (_Float16*)alloc(2048 * 64, 2);            // LN(sinc)
    float*    FW   = (float*)alloc((size_t)ROWS * 256, 4);
    float*    FS   = (float*)alloc(2048 * 256, 4);
    _Float16* HCAT = (_Float16*)alloc((size_t)ROWS * 512, 2);
    float*    FC   = (float*)alloc((size_t)ROWS * 256, 4);
    float*    SE   = (float*)alloc(BB * 256, 4);
    float*    RES  = (float*)alloc((size_t)ROWS * 256, 4);
    _Float16* XN   = (_Float16*)alloc((size_t)ROWS * 256, 2);
    float*    XZ   = (float*)alloc((size_t)ROWS * 1024, 4);
    _Float16* XC   = (_Float16*)alloc((size_t)2 * ROWS * DI, 2);
    float*    XDBL = (float*)alloc((size_t)2 * ROWS * 48, 4);
    _Float16* G    = (_Float16*)alloc((size_t)2 * ROWS * DI, 2);
    float*    MSUM = (float*)alloc((size_t)ROWS * 256, 4);
    _Float16* M16  = (_Float16*)alloc((size_t)ROWS * 256, 2);
    _Float16* H1   = HW;   // alias: HW dead after the W_wp GEMM

    auto cvt = [&](const float* s, _Float16* d, int n) {
        cvt_f32_to_f16<<<(n + 255) / 256, 256, 0, stream>>>(s, d, n);
    };
    cvt(W_wp, WHwp, 256 * 1024);
    cvt(W_sp, WHsp, 256 * 64);
    cvt(W_fuse, WHfuse, 256 * 512);
    cvt(W_in, WHin, 1024 * 256);
    cvt(W_xproj, WHxproj, 48 * 512);
    cvt(W_out, WHout, 256 * 512);
    cvt(W_ff1, WHff1, 1024 * 256);
    cvt(W_ff2, WHff2, 256 * 1024);

    auto gemm = [&](const _Float16* A, const _Float16* W, int M, int Nn, int K,
                    float* o32, _Float16* o16, int ldc, const float* bias,
                    const float* add, int mode) {
        int blocks = (M / 128) * (Nn / 16);
        wmma_gemm16<<<blocks, 256, 0, stream>>>(A, W, M, Nn, K, o32, o16, ldc, bias, add, mode);
    };

    // front-end
    ln_to_f16<<<ROWS, 256, 0, stream>>>(f_wavlm, ln_w_g, ln_w_b, HW, 1024);
    ln_to_f16<<<2048, 64, 0, stream>>>(f_sinc, ln_s_g, ln_s_b, HS, 64);
    gemm(HW, WHwp, ROWS, 256, 1024, FW, nullptr, 256, b_wp, nullptr, GM_F32);
    gemm(HS, WHsp, 2048, 256, 64, FS, nullptr, 256, b_sp, nullptr, GM_F32);
    hcat_kernel<<<ROWS, 256, 0, stream>>>(FW, FS, HCAT);
    gemm(HCAT, WHfuse, ROWS, 256, 512, FC, nullptr, 256, b_fuse, nullptr, GM_F32);
    se_kernel<<<BB, 256, 0, stream>>>(FC, W_se1, W_se2, SE);
    fuse_se_ln2_kernel<<<ROWS, 256, 0, stream>>>(FC, SE, ln_f_g, ln_f_b, n1_g, n1_b, RES, XN);

    // mamba (shared in-projection; per-direction conv/xproj/scan)
    gemm(XN, WHin, ROWS, 1024, 256, XZ, nullptr, 1024, nullptr, nullptr, GM_F32);
    conv_silu_kernel<<<dim3(ROWS, 2, 2), 256, 0, stream>>>(XZ, conv_w, conv_b, XC);
    gemm(XC, WHxproj, ROWS, 48, 512, XDBL, nullptr, 48, nullptr, nullptr, GM_F32);
    gemm(XC + (size_t)ROWS * DI, WHxproj, ROWS, 48, 512,
         XDBL + (size_t)ROWS * 48, nullptr, 48, nullptr, nullptr, GM_F32);
    scan_kernel<<<dim3(DI / 128, BB, 2), 128, 0, stream>>>(XDBL, XC, XZ, W_dt, b_dt, A_log, Dp, G);
    gemm(G, WHout, ROWS, 256, 512, MSUM, nullptr, 256, nullptr, nullptr, GM_F32);
    gemm(G + (size_t)ROWS * DI, WHout, ROWS, 256, 512, MSUM, nullptr, 256, nullptr, nullptr, GM_F32_ACC);
    ln_to_f16<<<ROWS, 256, 0, stream>>>(MSUM, n2_g, n2_b, M16, 256);

    // FFN + residual
    gemm(M16, WHff1, ROWS, 1024, 256, nullptr, H1, 1024, b_ff1, nullptr, GM_F16_GELU);
    gemm(H1, WHff2, ROWS, 256, 1024, (float*)d_out, nullptr, 256, b_ff2, RES, GM_F32_ADD);
}
